// SwitchGLU_43963285242757
// MI455X (gfx1250) — matrix operand order repaired and checked by reading, not compiled, not run on hardware
//
#include <hip/hip_runtime.h>
#include <hip/hip_bf16.h>

// Model dims
#define E_ 8
#define D_ 768
#define H_ 2048
#define B_ 2
#define S_ 512
#define K_ 2
#define N_ (B_*S_*K_)          // 2048 token-expert pairs
#define TM 32                  // rows per block tile
#define TNB 128                // cols per block tile (wave: 16x32, 8 waves -> 32x128)
#define MAXT (N_/TM + E_)      // 72 max row tiles across experts

typedef __attribute__((ext_vector_type(16))) __bf16          v16bf;
typedef __attribute__((ext_vector_type(16))) unsigned short  v16u;
typedef __attribute__((ext_vector_type(8)))  unsigned short  v8u;
typedef __attribute__((ext_vector_type(8)))  float           v8f;

#define V8F_ZERO {0.f,0.f,0.f,0.f,0.f,0.f,0.f,0.f}

#define WMMA_BF16(A, B, C) __builtin_amdgcn_wmma_f32_16x16x32_bf16( \
        false, (A), false, (B), (short)0, (C), false, false)

// scheduler pipeline directives (soft hints; no-op if unsupported)
#if __has_builtin(__builtin_amdgcn_sched_group_barrier)
#define SGB(mask, size) __builtin_amdgcn_sched_group_barrier((mask), (size), 0)
#else
#define SGB(mask, size)
#endif
#define SG_MFMA  0x8    // matches WMMA on gfx12 (isMFMAorWMMA)
#define SG_VMRD  0x20   // VMEM read

__device__ __forceinline__ unsigned short f2bf(float f) {
    unsigned int u = __float_as_uint(f);
    u += 0x7FFFu + ((u >> 16) & 1u);      // round-to-nearest-even
    return (unsigned short)(u >> 16);
}

// fast silu: g * sigmoid(g) with native v_exp_f32 + v_rcp_f32
__device__ __forceinline__ float fast_silu(float g) {
    float t = __expf(-g);
    return g * __builtin_amdgcn_rcpf(1.0f + t);
}

// ---------------- fp32 -> bf16 bulk convert (weights) ----------------
__global__ void cvt_bf16_kernel(const float* __restrict__ src,
                                unsigned short* __restrict__ dst, int n4) {
    int i = blockIdx.x * blockDim.x + threadIdx.x;
    int stride = gridDim.x * blockDim.x;
    const float4* s4 = (const float4*)src;
    ushort4* d4 = (ushort4*)dst;
    for (; i < n4; i += stride) {
        float4 v = s4[i];
        ushort4 o;
        o.x = f2bf(v.x); o.y = f2bf(v.y); o.z = f2bf(v.z); o.w = f2bf(v.w);
        d4[i] = o;
    }
}

// ---------------- dispatch: counts, tile map, sorted row map ----------------
__global__ void setup_kernel(const int* __restrict__ idx,
                             int* __restrict__ row_map,
                             int* __restrict__ tmeta /* [1 + 3*MAXT] */) {
    __shared__ int cnt[E_], offs[E_], rank[E_];
    int t = threadIdx.x;
    if (t < E_) { cnt[t] = 0; rank[t] = 0; }
    __syncthreads();
    for (int n = t; n < N_; n += blockDim.x) atomicAdd(&cnt[idx[n]], 1);
    __syncthreads();
    if (t == 0) {
        int off = 0, tt = 0;
        for (int e = 0; e < E_; ++e) {
            offs[e] = off;
            int c = cnt[e];
            int nt = (c + TM - 1) / TM;
            for (int i = 0; i < nt; ++i) {
                int r0 = off + TM * i;
                int r1 = off + TM * (i + 1);
                int end = off + c;
                tmeta[1 + 3 * tt + 0] = e;
                tmeta[1 + 3 * tt + 1] = r0;
                tmeta[1 + 3 * tt + 2] = (r1 < end) ? r1 : end;
                ++tt;
            }
            off += c;
        }
        tmeta[0] = tt;
    }
    __syncthreads();
    for (int n = t; n < N_; n += blockDim.x) {
        int e = idx[n];
        int p = offs[e] + atomicAdd(&rank[e], 1);
        row_map[p] = n;   // sorted position -> pair index
    }
}

// ---------------- gather token rows into sorted bf16 activation buffer ----------------
__global__ void gather_kernel(const float* __restrict__ x,
                              const int* __restrict__ row_map,
                              unsigned short* __restrict__ xs) {
    int pos = blockIdx.x;
    int n = row_map[pos];
    int tok = n / K_;
    const float4* src = (const float4*)(x + (size_t)tok * D_);
    ushort4* dst = (ushort4*)(xs + (size_t)pos * D_);
    for (int d = threadIdx.x; d < D_ / 4; d += blockDim.x) {
        float4 v = src[d];
        ushort4 o;
        o.x = f2bf(v.x); o.y = f2bf(v.y); o.z = f2bf(v.z); o.w = f2bf(v.w);
        dst[d] = o;
    }
}

// helper: build a 16x32 bf16 A-fragment from a row-major bf16 row pointer
__device__ __forceinline__ v16bf load_a_frag(const unsigned short* rowptr, int aoff) {
    v8u a0 = *(const v8u*)(rowptr + aoff);
    v8u a1 = *(const v8u*)(rowptr + aoff + 8);
    v16u au = __builtin_shufflevector(a0, a1,
        0,1,2,3,4,5,6,7,8,9,10,11,12,13,14,15);
    return __builtin_bit_cast(v16bf, au);
}

// helper: B fragment = 16 contiguous K values of one weight row (column of B)
__device__ __forceinline__ v16bf load_b_frag(const unsigned short* colptr) {
    v16u bu = *(const v16u*)colptr;
    return __builtin_bit_cast(v16bf, bu);
}

// ---------------- GEMM1: h = silu(x Wg^T) * (x Wu^T), grouped by expert ----------------
// block tile 32 rows x 128 cols; wave tile 16 rows x 32 cols (gate+up => 4 wmma/k-step)
// fully unrolled; sched_group_barrier forces a 2-iteration load lookahead pipeline
__global__ __launch_bounds__(256)
void gemm1_swiglu_kernel(const unsigned short* __restrict__ xs,
                         const unsigned short* __restrict__ wg,
                         const unsigned short* __restrict__ wu,
                         unsigned short* __restrict__ h,
                         const int* __restrict__ tmeta) {
    int nT = tmeta[0];
    int tileId = blockIdx.x;
    if (tileId >= nT) return;                       // block-uniform exit
    int e  = tmeta[1 + 3 * tileId + 0];
    int r0 = tmeta[1 + 3 * tileId + 1];
    int r1 = tmeta[1 + 3 * tileId + 2];

    int lane = threadIdx.x & 31;
    int wid  = threadIdx.x >> 5;
    int wr = wid & 1;          // 2 row sub-tiles of 16
    int wc = wid >> 1;         // 4 col groups of 32

    // A fragment addressing (16x32 bf16, ISA 7.12.2)
    int mrow = (lane & 15) + wr * 16;
    int arow = r0 + mrow; if (arow > r1 - 1) arow = r1 - 1;   // clamp padding rows
    const unsigned short* aptr = xs + (size_t)arow * D_;
    int aoff = (lane >> 4) * 16;

    // B fragments: two 16-col strips per wave, for gate and up
    int col0 = blockIdx.y * TNB + wc * 32 + (lane & 15);      // H index
    int boff = (lane >> 4) * 16;
    const unsigned short* gptr0 = wg + ((size_t)e * H_ + col0) * D_ + boff;
    const unsigned short* gptr1 = gptr0 + (size_t)16 * D_;
    const unsigned short* uptr0 = wu + ((size_t)e * H_ + col0) * D_ + boff;
    const unsigned short* uptr1 = uptr0 + (size_t)16 * D_;

    v8f accg0 = V8F_ZERO, accg1 = V8F_ZERO;
    v8f accu0 = V8F_ZERO, accu1 = V8F_ZERO;

#pragma unroll
    for (int k0 = 0; k0 < D_; k0 += 32) {   // 24 iterations, fully unrolled
        v16bf av = load_a_frag(aptr + k0, aoff);
        v16bf g0 = load_b_frag(gptr0 + k0);
        v16bf g1 = load_b_frag(gptr1 + k0);
        v16bf u0 = load_b_frag(uptr0 + k0);
        v16bf u1 = load_b_frag(uptr1 + k0);

        accg0 = WMMA_BF16(av, g0, accg0);
        accg1 = WMMA_BF16(av, g1, accg1);
        accu0 = WMMA_BF16(av, u0, accu0);
        accu1 = WMMA_BF16(av, u1, accu1);
    }

    // pipeline: V,V,(M,V)x22,M,M  -> each WMMA group waits only on loads
    // issued 2 groups earlier; ~20 loads stay in flight during WMMA execution
    SGB(SG_VMRD, 10); SGB(SG_VMRD, 10);
#pragma unroll
    for (int i = 0; i < 22; ++i) { SGB(SG_MFMA, 4); SGB(SG_VMRD, 10); }
    SGB(SG_MFMA, 4); SGB(SG_MFMA, 4);

    // C layout: element r -> row (lane>>4)*8 + r, col lane&15
    int mBase = (lane >> 4) * 8;
#pragma unroll
    for (int r = 0; r < 8; ++r) {
        int row = r0 + wr * 16 + mBase + r;
        if (row < r1) {
            unsigned short* hp = h + (size_t)row * H_ + col0;
            hp[0]  = f2bf(fast_silu(accg0[r]) * accu0[r]);
            hp[16] = f2bf(fast_silu(accg1[r]) * accu1[r]);
        }
    }
}

// ---------------- GEMM2: out = h Wd^T, scatter rows back to pair order ----------------
// block tile 32 rows x 128 cols; wave tile 16 rows x 32 cols; fully unrolled,
// lag-3 load lookahead (WMMA groups are half-size here)
__global__ __launch_bounds__(256)
void gemm2_scatter_kernel(const unsigned short* __restrict__ h,
                          const unsigned short* __restrict__ wd,
                          const int* __restrict__ row_map,
                          const int* __restrict__ tmeta,
                          float* __restrict__ out) {
    int nT = tmeta[0];
    int tileId = blockIdx.x;
    if (tileId >= nT) return;
    int e  = tmeta[1 + 3 * tileId + 0];
    int r0 = tmeta[1 + 3 * tileId + 1];
    int r1 = tmeta[1 + 3 * tileId + 2];

    int lane = threadIdx.x & 31;
    int wid  = threadIdx.x >> 5;
    int wr = wid & 1;
    int wc = wid >> 1;

    int mrow = (lane & 15) + wr * 16;
    int arow = r0 + mrow; if (arow > r1 - 1) arow = r1 - 1;
    const unsigned short* aptr = h + (size_t)arow * H_;
    int aoff = (lane >> 4) * 16;

    int col0 = blockIdx.y * TNB + wc * 32 + (lane & 15);      // D index
    int boff = (lane >> 4) * 16;
    const unsigned short* bptr0 = wd + ((size_t)e * D_ + col0) * H_ + boff;
    const unsigned short* bptr1 = bptr0 + (size_t)16 * H_;

    v8f acc0 = V8F_ZERO, acc1 = V8F_ZERO;

#pragma unroll
    for (int k0 = 0; k0 < H_; k0 += 32) {   // 64 iterations, fully unrolled
        v16bf av = load_a_frag(aptr + k0, aoff);
        v16bf b0 = load_b_frag(bptr0 + k0);
        v16bf b1 = load_b_frag(bptr1 + k0);

        acc0 = WMMA_BF16(av, b0, acc0);
        acc1 = WMMA_BF16(av, b1, acc1);
    }

    // pipeline: V,V,V,(M,V)x61,M,M,M
    SGB(SG_VMRD, 6); SGB(SG_VMRD, 6); SGB(SG_VMRD, 6);
#pragma unroll
    for (int i = 0; i < 61; ++i) { SGB(SG_MFMA, 2); SGB(SG_VMRD, 6); }
    SGB(SG_MFMA, 2); SGB(SG_MFMA, 2); SGB(SG_MFMA, 2);

    int mBase = (lane >> 4) * 8;
#pragma unroll
    for (int r = 0; r < 8; ++r) {
        int row = r0 + wr * 16 + mBase + r;
        if (row < r1) {
            int n = row_map[row];                  // pair index
            float* op = out + (size_t)n * D_ + col0;
            op[0]  = acc0[r];
            op[16] = acc1[r];
        }
    }
}

// ---------------- host side ----------------
extern "C" void kernel_launch(void* const* d_in, const int* in_sizes, int n_in,
                              void* d_out, int out_size, void* d_ws, size_t ws_size,
                              hipStream_t stream) {
    const float* x      = (const float*)d_in[0];   // (B,S,D)
    const int*   idx    = (const int*)d_in[1];     // (B,S,K)
    const float* w_gate = (const float*)d_in[2];   // (E,H,D)
    const float* w_up   = (const float*)d_in[3];   // (E,H,D)
    const float* w_down = (const float*)d_in[4];   // (E,D,H)
    float* out = (float*)d_out;                    // (B,S,K,D)

    const size_t WELEMS = (size_t)E_ * H_ * D_;    // same for all three

    // workspace carve-out (256B aligned)
    char* p = (char*)d_ws;
    auto carve = [&](size_t bytes) -> char* {
        char* r = p; p += (bytes + 255) & ~(size_t)255; return r;
    };
    unsigned short* wg_bf  = (unsigned short*)carve(WELEMS * 2);
    unsigned short* wu_bf  = (unsigned short*)carve(WELEMS * 2);
    unsigned short* wd_bf  = (unsigned short*)carve(WELEMS * 2);
    unsigned short* xs     = (unsigned short*)carve((size_t)N_ * D_ * 2);
    unsigned short* hbuf   = (unsigned short*)carve((size_t)N_ * H_ * 2);
    int* row_map           = (int*)carve((size_t)N_ * 4);
    int* tmeta             = (int*)carve((size_t)(1 + 3 * MAXT) * 4);
    (void)ws_size; (void)in_sizes; (void)n_in; (void)out_size;

    int n4 = (int)(WELEMS / 4);
    cvt_bf16_kernel<<<4096, 256, 0, stream>>>(w_gate, wg_bf, n4);
    cvt_bf16_kernel<<<4096, 256, 0, stream>>>(w_up,   wu_bf, n4);
    cvt_bf16_kernel<<<4096, 256, 0, stream>>>(w_down, wd_bf, n4);

    setup_kernel<<<1, 256, 0, stream>>>(idx, row_map, tmeta);
    gather_kernel<<<N_, 256, 0, stream>>>(x, row_map, xs);

    dim3 g1(MAXT, H_ / TNB);   // 72 x 16
    gemm1_swiglu_kernel<<<g1, 256, 0, stream>>>(xs, wg_bf, wu_bf, hbuf, tmeta);

    dim3 g2(MAXT, D_ / TNB);   // 72 x 6
    gemm2_scatter_kernel<<<g2, 256, 0, stream>>>(hbuf, wd_bf, row_map, tmeta, out);
}
